// SparseGeomAttention_16088947490884
// MI455X (gfx1250) — compile-verified, matrix operand order checked by reading, not run on hardware
//
#include <hip/hip_runtime.h>
#include <hip/hip_bf16.h>
#include <math.h>

// ---------------------------------------------------------------------------
// SparseGeomAttention for MI455X (gfx1250, wave32, WMMA)
//
// D_MODEL=512, N_HEADS=8, D_K=64, B=8192, K=32, M=16
//
//   1) pack_weights : W_{q,k,v,o} f32 -> f16 WMMA B-fragment layout
//   2) gemm_rows    : Q = x_anc @ Wq        (v_wmma_f32_16x16x32_f16)
//   3) fused_attn   : per-anchor K/V projection (WMMA) + graph-biased softmax
//                     attention, all in LDS; x_nei (512MB) read exactly once,
//                     staged with GLOBAL_LOAD_ASYNC_TO_LDS_B128 when available
//   4) gemm_rows    : out = ctx @ Wo
//
// GEMM inner loops are ks-outermost with 4 live N-tile accumulators, single
// A-fragment load per ks (reused 4x) and double-buffered B fragments, so the
// L2 latency of the B loads overlaps with 8 WMMAs of compute and only ~150
// VGPRs are live (no spills).
// ---------------------------------------------------------------------------

typedef __attribute__((ext_vector_type(16))) _Float16 v16h;
typedef __attribute__((ext_vector_type(8)))  float    v8f;
typedef __attribute__((ext_vector_type(4)))  int      v4i;

#define DM   512
#define NH   8
#define DK   64
#define NB   8192
#define KN   32
#define MM   16
#define KVSTRIDE 513   // 32 x 513 f32; stride 513 => conflict-free column walks

// Per-weight packed fragment count: 32 ntiles * 16 ksteps * 32 lanes * 16 halves
#define WFRAG_ELEMS (32 * 16 * 32 * 16)   // 262144 halves = 512KB

// -------- async global->LDS staging (CDNA5 ASYNCcnt path), with fallback ----
#if defined(__gfx1250__) && \
    __has_builtin(__builtin_amdgcn_global_load_async_to_lds_b128) && \
    __has_builtin(__builtin_amdgcn_s_wait_asynccnt)
#define USE_ASYNC_LDS 1
#else
#define USE_ASYNC_LDS 0
#endif

#if USE_ASYNC_LDS
__device__ __forceinline__ void async_copy_b128(const void* g, void* l) {
    __builtin_amdgcn_global_load_async_to_lds_b128(
        (__attribute__((address_space(1))) v4i*)g,
        (__attribute__((address_space(3))) v4i*)l,
        /*imm offset=*/0, /*cpol=*/0);
}
__device__ __forceinline__ void wait_async0() {
    __builtin_amdgcn_s_wait_asynccnt(0);
}
#endif

// 16-bit A-matrix 16x32 fragment layout (ISA 7.12.2):
//   lanes 0-15  (M = lane)    : halves j=0..7 -> K=j,    j=8..15 -> K=j+8
//   lanes 16-31 (M = lane-16) : halves j=0..7 -> K=j+8,  j=8..15 -> K=j+16
// B (32x16, KxN) uses the mirrored map with N = lane%16.
__device__ __forceinline__ int frag_k(int lane, int j) {
    return (lane < 16) ? ((j < 8) ? j : j + 8)
                       : ((j < 8) ? j + 8 : j + 16);
}

__device__ __forceinline__ v8f wmma_f16(v16h a, v16h b, v8f c) {
    return __builtin_amdgcn_wmma_f32_16x16x32_f16(
        /*neg_a=*/false, a, /*neg_b=*/false, b,
        /*c_mod=*/(short)0, c, /*reuse_a=*/false, /*reuse_b=*/false);
}

// ---------------------------------------------------------------------------
// 1) Pack four 512x512 f32 weights into f16 WMMA B-fragment layout.
//    dst[w][nt][ks][lane][j] = (f16) W_w[ ks*32 + frag_k(lane,j) ][ nt*16 + lane%16 ]
// ---------------------------------------------------------------------------
__global__ void pack_weights_kernel(const float* __restrict__ Wq,
                                    const float* __restrict__ Wk,
                                    const float* __restrict__ Wv,
                                    const float* __restrict__ Wo,
                                    _Float16* __restrict__ Wpack) {
    int idx = blockIdx.x * blockDim.x + threadIdx.x;   // 4 * 2^18 total
    int j    =  idx        & 15;
    int lane = (idx >>  4) & 31;
    int ks   = (idx >>  9) & 15;
    int nt   = (idx >> 13) & 31;
    int w    =  idx >> 18;
    const float* W = (w == 0) ? Wq : (w == 1) ? Wk : (w == 2) ? Wv : Wo;
    int k = ks * 32 + frag_k(lane, j);
    int n = nt * 16 + (lane & 15);
    Wpack[idx] = (_Float16)W[k * DM + n];
}

// ---------------------------------------------------------------------------
// 2/4) Row-tile GEMM: Out[16 rows, 512] = X[16 rows, 512] @ W(packed)
//      One block per 16-row tile; 8 waves; each wave keeps 4 N-tile
//      accumulators live across the ks loop (max B reuse of each A fragment).
//      Dynamic LDS: A fragments 16KB + raw X tile 32KB.
// ---------------------------------------------------------------------------
__global__ void gemm_rows_kernel(const float* __restrict__ X,
                                 const _Float16* __restrict__ Wpack,
                                 float* __restrict__ Out) {
    extern __shared__ __align__(32) char smem[];
    _Float16* Astage = (_Float16*)smem;                    // 16*32*16 halves = 16KB
    float*    raw    = (float*)(smem + 16384);             // 16*512 f32     = 32KB

    const int tid  = threadIdx.x;
    const int lane = tid & 31;
    const int w    = tid >> 5;
    const size_t row0 = (size_t)blockIdx.x * 16;
    const float* Xb = X + row0 * DM;

#if USE_ASYNC_LDS
    // Stage raw 32KB tile with async global->LDS b128 (8 per thread).
    for (int ofs = tid * 16; ofs < 16 * DM * 4; ofs += 256 * 16)
        async_copy_b128((const char*)Xb + ofs, (char*)raw + ofs);
    wait_async0();
    __syncthreads();
    // LDS->LDS f32 -> f16 fragment shuffle.
    for (int e = tid; e < 16 * 32 * 16; e += 256) {
        int j  =  e       & 15;
        int ln = (e >> 4) & 31;
        int ks =  e >> 9;
        Astage[e] = (_Float16)raw[(ln & 15) * DM + ks * 32 + frag_k(ln, j)];
    }
#else
    for (int e = tid; e < 16 * 32 * 16; e += 256) {
        int j  =  e       & 15;
        int ln = (e >> 4) & 31;
        int ks =  e >> 9;
        Astage[e] = (_Float16)Xb[(ln & 15) * DM + ks * 32 + frag_k(ln, j)];
    }
#endif
    __syncthreads();

    const v16h* Bp = (const v16h*)Wpack;
    v8f zero = {};
    v8f acc[4];
#pragma unroll
    for (int i = 0; i < 4; ++i) acc[i] = zero;

    v16h bcur[4];
#pragma unroll
    for (int i = 0; i < 4; ++i) bcur[i] = Bp[((w + 8 * i) * 16 + 0) * 32 + lane];

#pragma unroll
    for (int ks = 0; ks < 16; ++ks) {
        v16h bnxt[4];
#pragma unroll
        for (int i = 0; i < 4; ++i)
            bnxt[i] = (ks + 1 < 16) ? Bp[((w + 8 * i) * 16 + ks + 1) * 32 + lane]
                                    : bcur[i];
        v16h a = *(const v16h*)&Astage[(ks * 32 + lane) * 16];
#pragma unroll
        for (int i = 0; i < 4; ++i) acc[i] = wmma_f16(a, bcur[i], acc[i]);
#pragma unroll
        for (int i = 0; i < 4; ++i) bcur[i] = bnxt[i];
    }

    // C/D layout: VGPR j -> (M = j or j+8, N = lane%16)
    const int mbase = (lane < 16) ? 0 : 8;
#pragma unroll
    for (int i = 0; i < 4; ++i) {
        int n = (w + 8 * i) * 16 + (lane & 15);
#pragma unroll
        for (int j = 0; j < 8; ++j)
            Out[(row0 + mbase + j) * DM + n] = acc[i][j];
    }
}

// ---------------------------------------------------------------------------
// 3) Fused per-anchor K/V projection + geometry-biased attention.
//    LDS: Astage 32KB | KV 32*513*4 = 64.1KB (raw x_nei tile aliases here)
//         | attn 1KB  => ~97.4KB dynamic
// ---------------------------------------------------------------------------
__device__ __forceinline__ void project_32x512(const _Float16* __restrict__ Astage,
                                               const _Float16* __restrict__ Wpack,
                                               float* __restrict__ KV,
                                               int lane, int w) {
    const v16h* Bp = (const v16h*)Wpack;
    v8f zero = {};
    v8f acc0[4], acc1[4];
#pragma unroll
    for (int i = 0; i < 4; ++i) { acc0[i] = zero; acc1[i] = zero; }

    v16h bcur[4];
#pragma unroll
    for (int i = 0; i < 4; ++i) bcur[i] = Bp[((w + 8 * i) * 16 + 0) * 32 + lane];

#pragma unroll
    for (int ks = 0; ks < 16; ++ks) {
        v16h bnxt[4];
#pragma unroll
        for (int i = 0; i < 4; ++i)
            bnxt[i] = (ks + 1 < 16) ? Bp[((w + 8 * i) * 16 + ks + 1) * 32 + lane]
                                    : bcur[i];
        v16h a0 = *(const v16h*)&Astage[((0 * 16 + ks) * 32 + lane) * 16];
        v16h a1 = *(const v16h*)&Astage[((1 * 16 + ks) * 32 + lane) * 16];
#pragma unroll
        for (int i = 0; i < 4; ++i) {
            acc0[i] = wmma_f16(a0, bcur[i], acc0[i]);
            acc1[i] = wmma_f16(a1, bcur[i], acc1[i]);
        }
#pragma unroll
        for (int i = 0; i < 4; ++i) bcur[i] = bnxt[i];
    }

    const int mbase = (lane < 16) ? 0 : 8;
#pragma unroll
    for (int i = 0; i < 4; ++i) {
        int n = (w + 8 * i) * 16 + (lane & 15);
#pragma unroll
        for (int j = 0; j < 8; ++j) {
            KV[(mbase + j)      * KVSTRIDE + n] = acc0[i][j];
            KV[(16 + mbase + j) * KVSTRIDE + n] = acc1[i][j];
        }
    }
}

__global__ void fused_attn_kernel(const float* __restrict__ x_nei,
                                  const float* __restrict__ pi,
                                  const float* __restrict__ tau,
                                  const float* __restrict__ ew,
                                  const _Float16* __restrict__ WkP,
                                  const _Float16* __restrict__ WvP,
                                  const float* __restrict__ Qbuf,
                                  float* __restrict__ Ctx) {
    extern __shared__ __align__(32) char smem[];
    _Float16* Astage = (_Float16*)smem;                                 // 32KB
    float*    KV     = (float*)(smem + 32768);                          // 32*513 f32
    float*    attn   = (float*)(smem + 32768 + KN * KVSTRIDE * 4);      // 8*32 f32

    const int tid  = threadIdx.x;
    const int lane = tid & 31;
    const int w    = tid >> 5;
    const size_t b = blockIdx.x;
    const float* Xb = x_nei + b * KN * DM;   // 32x512 f32 = 64KB contiguous

#if USE_ASYNC_LDS
    // Async-stage the raw 64KB tile into the (currently unused) KV region.
    float* rawt = KV;
    for (int ofs = tid * 16; ofs < KN * DM * 4; ofs += 256 * 16)
        async_copy_b128((const char*)Xb + ofs, (char*)rawt + ofs);
    wait_async0();
    __syncthreads();
    // LDS->LDS f32 -> f16 A-fragment shuffle (2 M-tiles).
    for (int e = tid; e < 2 * 16 * 32 * 16; e += 256) {
        int j  =  e        & 15;
        int ln = (e >>  4) & 31;
        int ks = (e >>  9) & 15;
        int mt =  e >> 13;
        Astage[e] = (_Float16)rawt[(mt * 16 + (ln & 15)) * DM + ks * 32 + frag_k(ln, j)];
    }
#else
    for (int e = tid; e < 2 * 16 * 32 * 16; e += 256) {
        int j  =  e        & 15;
        int ln = (e >>  4) & 31;
        int ks = (e >>  9) & 15;
        int mt =  e >> 13;
        Astage[e] = (_Float16)Xb[(mt * 16 + (ln & 15)) * DM + ks * 32 + frag_k(ln, j)];
    }
#endif
    __syncthreads();

    // ---- K projection: Kp[32][512] -> KV (overwrites raw tile) ----
    project_32x512(Astage, WkP, KV, lane, w);
    __syncthreads();

    // ---- Attention: wave w == head h; lane == neighbor k ----
    {
        const int h = w;
        const float* pib = pi + (b * NH + h) * MM;
        const float* ewb = ew + (b * KN + lane) * MM;
        float aeff = 0.f;
#pragma unroll
        for (int m = 0; m < MM; ++m) aeff += pib[m] * ewb[m];
        float bias = logf(tau[h] * aeff + 1e-6f);

        const float* qb = Qbuf + b * DM + h * DK;
        const float* kr = KV + lane * KVSTRIDE + h * DK;
        float dot = 0.f;
#pragma unroll
        for (int d = 0; d < DK; ++d) dot += qb[d] * kr[d];
        float s = dot * 0.125f + bias;

        float mx = s;
#pragma unroll
        for (int off = 16; off > 0; off >>= 1) mx = fmaxf(mx, __shfl_xor(mx, off, 32));
        float e = __expf(s - mx);
        float sum = e;
#pragma unroll
        for (int off = 16; off > 0; off >>= 1) sum += __shfl_xor(sum, off, 32);
        attn[h * KN + lane] = e / sum;
    }
    __syncthreads();   // all reads of Kp done before overwrite

    // ---- V projection reuses the same KV buffer ----
    project_32x512(Astage, WvP, KV, lane, w);
    __syncthreads();

    // ---- ctx[h, d] = sum_k attn[h,k] * Vp[k, h*64+d]; lane owns 2 channels ----
    {
        const int h   = w;
        const int col = h * DK + lane * 2;
        float c0 = 0.f, c1 = 0.f;
#pragma unroll
        for (int k = 0; k < KN; ++k) {
            float a = attn[h * KN + k];
            c0 = fmaf(a, KV[k * KVSTRIDE + col],     c0);
            c1 = fmaf(a, KV[k * KVSTRIDE + col + 1], c1);
        }
        Ctx[b * DM + col]     = c0;
        Ctx[b * DM + col + 1] = c1;
    }
}

// ---------------------------------------------------------------------------
// Launch: pack -> Q gemm -> fused attention -> output gemm
// Workspace layout: [Wpack 2MB][Qbuf 16MB][Ctx 16MB]
// ---------------------------------------------------------------------------
extern "C" void kernel_launch(void* const* d_in, const int* in_sizes, int n_in,
                              void* d_out, int out_size, void* d_ws, size_t ws_size,
                              hipStream_t stream) {
    const float* x_anc = (const float*)d_in[0];
    const float* x_nei = (const float*)d_in[1];
    const float* pi    = (const float*)d_in[2];
    const float* tau   = (const float*)d_in[3];
    const float* ew    = (const float*)d_in[4];
    const float* Wq    = (const float*)d_in[5];
    const float* Wk    = (const float*)d_in[6];
    const float* Wv    = (const float*)d_in[7];
    const float* Wo    = (const float*)d_in[8];
    float* out = (float*)d_out;

    char* ws = (char*)d_ws;
    _Float16* Wpack = (_Float16*)ws;                                   // 4 * 512KB
    float*    Qbuf  = (float*)(ws + (size_t)4 * WFRAG_ELEMS * 2);      // 16MB
    float*    Ctx   = (float*)(ws + (size_t)4 * WFRAG_ELEMS * 2
                                  + (size_t)NB * DM * 4);              // 16MB

    pack_weights_kernel<<<(4 * WFRAG_ELEMS) / 256, 256, 0, stream>>>(Wq, Wk, Wv, Wo, Wpack);

    size_t sh_gemm = 16384 + 16 * DM * 4;                              // 48KB
    gemm_rows_kernel<<<NB / 16, 256, sh_gemm, stream>>>(x_anc, Wpack + 0 * WFRAG_ELEMS, Qbuf);

    size_t sh_attn = 32768 + (size_t)KN * KVSTRIDE * 4 + (size_t)NH * KN * 4;  // ~97.4KB
    fused_attn_kernel<<<NB, 256, sh_attn, stream>>>(x_nei, pi, tau, ew,
                                                    Wpack + 1 * WFRAG_ELEMS,
                                                    Wpack + 2 * WFRAG_ELEMS,
                                                    Qbuf, Ctx);

    gemm_rows_kernel<<<NB / 16, 256, sh_gemm, stream>>>(Ctx, Wpack + 3 * WFRAG_ELEMS, out);
}
